// MultiHeadAttention_35837207118223
// MI455X (gfx1250) — matrix operand order repaired and checked by reading, not compiled, not run on hardware
//
#include <hip/hip_runtime.h>

#define BB 16
#define SS 1024
#define DD 512
#define HH 8
#define DEPTH 64

typedef __attribute__((ext_vector_type(16))) __bf16 v16bf;
typedef __attribute__((ext_vector_type(8)))  float  v8f;

// ---------------------------------------------------------------------------
// WMMA helper: D = A(16x32 bf16) x B(32x16 bf16) + C(16x16 f32)
// ---------------------------------------------------------------------------
__device__ __forceinline__ v8f wmma_bf16(v16bf a, v16bf b, v8f c) {
  return __builtin_amdgcn_wmma_f32_16x16x32_bf16(
      /*neg_a=*/false, a, /*neg_b=*/false, b,
      /*c_mod=*/(short)0, c, /*reuse_a=*/false, /*reuse_b=*/false);
}

// ---------------------------------------------------------------------------
// Async global->LDS copy (16B per lane), tracked with ASYNCcnt.
// Low 32 bits of a flat shared-aperture address are the LDS byte offset.
// ---------------------------------------------------------------------------
__device__ __forceinline__ void async_copy_b128(void* lds_ptr, const void* gptr) {
  unsigned lds_off = (unsigned)(unsigned long long)lds_ptr;
  asm volatile("global_load_async_to_lds_b128 %0, %1, off"
               :: "v"(lds_off), "v"(gptr) : "memory");
}
__device__ __forceinline__ void wait_async_all() {
  asm volatile("s_wait_asynccnt 0x0" ::: "memory");
}

// A-matrix 16x32 (MxK) bf16 fragment from LDS tile with row stride `ld`.
// Lanes 0-15 hold M=lane, K={kb+0..7, 16+kb+0..7} with kb = (lane>>4)*8
__device__ __forceinline__ void load_a_frag(const __bf16* lds, int ld, int lane,
                                            v16bf& a) {
  const int row = lane & 15;
  const int kb  = (lane >> 4) << 3;  // 0 or 8
#pragma unroll
  for (int i = 0; i < 8; ++i) {
    const int k = ((i & 4) << 2) + kb + ((i & 3) << 1);
    a[2 * i]     = lds[row * ld + k];
    a[2 * i + 1] = lds[row * ld + k + 1];
  }
}

// B-matrix 32x16 (KxN) bf16 fragment from LDS tile stored N-major: lds[n*ld + k].
// Lane n%16 holds column N=n%16; K={kb+0..7, 16+kb+0..7} with kb = (lane>>4)*8
__device__ __forceinline__ void load_b_frag(const __bf16* lds, int ld, int lane,
                                            v16bf& b) {
  const int n  = lane & 15;
  const int kb = (lane >> 4) << 3;  // 0 or 8
#pragma unroll
  for (int j = 0; j < 8; ++j) {
    const int k = ((j & 4) << 2) + kb + ((j & 3) << 1);
    b[2 * j]     = lds[n * ld + k];
    b[2 * j + 1] = lds[n * ld + k + 1];
  }
}

// C/D 16x16 f32: N = lane&15, M = ((lane>>4)<<3) + r for acc[r]

// ---------------------------------------------------------------------------
// Kernel 1: Y = (X @ W^T + bias), head-split bf16 output with caller-chosen
// strides so Q/K use [b,h,s,d] and V uses [b,h,d,s] with NO branches:
//   element (h, s, d) stored at  h*(S*DEPTH) + s*s_stride + d*d_stride
// Block: 256 threads = 8 waves. Tile: 128 rows x 64 cols. K = 512.
// ---------------------------------------------------------------------------
__global__ __launch_bounds__(256) void proj_qkv_kernel(
    const float* __restrict__ X,     // [B*S, D] fp32
    const float* __restrict__ W,     // [D, D] fp32 (row = out feature)
    const float* __restrict__ bias,  // [D]
    __bf16* __restrict__ Y,
    int s_stride, int d_stride)
{
  __shared__ __attribute__((aligned(16))) __bf16 sA[128 * 32];
  __shared__ __attribute__((aligned(16))) __bf16 sB[64 * 32];
  const int tid  = threadIdx.x;
  const int wave = tid >> 5;
  const int lane = tid & 31;
  const int m0 = blockIdx.x * 128;
  const int n0 = blockIdx.y * 64;

  v8f acc[4];
#pragma unroll
  for (int t = 0; t < 4; ++t)
#pragma unroll
    for (int r = 0; r < 8; ++r) acc[t][r] = 0.f;

  for (int k0 = 0; k0 < DD; k0 += 32) {
    for (int i = tid; i < 128 * 32; i += 256) {
      const int r = i >> 5, c = i & 31;
      sA[i] = (__bf16)X[(size_t)(m0 + r) * DD + k0 + c];
    }
    for (int i = tid; i < 64 * 32; i += 256) {
      const int n = i >> 5, c = i & 31;
      sB[i] = (__bf16)W[(size_t)(n0 + n) * DD + k0 + c];
    }
    __syncthreads();
    v16bf a, b0, b1, b2, b3;
    load_a_frag(sA + wave * 16 * 32, 32, lane, a);
    load_b_frag(sB + 0 * 16 * 32, 32, lane, b0);
    load_b_frag(sB + 1 * 16 * 32, 32, lane, b1);
    load_b_frag(sB + 2 * 16 * 32, 32, lane, b2);
    load_b_frag(sB + 3 * 16 * 32, 32, lane, b3);
    acc[0] = wmma_bf16(a, b0, acc[0]);
    acc[1] = wmma_bf16(a, b1, acc[1]);
    acc[2] = wmma_bf16(a, b2, acc[2]);
    acc[3] = wmma_bf16(a, b3, acc[3]);
    __syncthreads();
  }

  const int ncol  = lane & 15;
  const int mbase = (lane >> 4) << 3;
#pragma unroll
  for (int t = 0; t < 4; ++t) {
    const int n = n0 + t * 16 + ncol;
    const int h = n >> 6, d = n & 63;
    const float bv = bias[n];
    const size_t dpart = (size_t)h * (SS * DEPTH) + (size_t)d * d_stride;
#pragma unroll
    for (int r = 0; r < 8; ++r) {
      const int m = m0 + wave * 16 + mbase + r;
      const int b = m >> 10, s = m & 1023;
      Y[(size_t)b * (HH * SS * DEPTH) + dpart + (size_t)s * s_stride] =
          (__bf16)(acc[t][r] + bv);
    }
  }
}

// ---------------------------------------------------------------------------
// Kernel 2: logits[b,h,q,k] = (Qh . Kh)/8 + mask[b,k] * -1e9   (fp32 -> d_out)
// Q/K tiles staged via async global->LDS b128 copies (bf16, no conversion).
// ---------------------------------------------------------------------------
__global__ __launch_bounds__(256) void logits_kernel(
    const __bf16* __restrict__ Qh,   // [B,H,S,DEPTH]
    const __bf16* __restrict__ Kh,   // [B,H,S,DEPTH]
    const int* __restrict__ mask,    // [B,1,1,S]
    float* __restrict__ attn)        // [B,H,S,S]
{
  __shared__ __attribute__((aligned(16))) __bf16 sQ[128 * 64];
  __shared__ __attribute__((aligned(16))) __bf16 sK[64 * 64];
  const int tid  = threadIdx.x;
  const int wave = tid >> 5;
  const int lane = tid & 31;
  const int q0 = blockIdx.x * 128;
  const int k0 = blockIdx.y * 64;
  const int bh = blockIdx.z;         // b*H + h
  const int b  = bh >> 3;
  const size_t base = (size_t)bh * SS * DEPTH;

  // sQ: 128 rows x 64 bf16 = 1024 chunks of 16B ; row = i>>3, colchunk = i&7
  for (int i = tid; i < 1024; i += 256) {
    const int r = i >> 3, c = (i & 7) << 3;
    async_copy_b128(&sQ[r * 64 + c], &Qh[base + (size_t)(q0 + r) * DEPTH + c]);
  }
  // sK: 64 rows x 64 bf16 = 512 chunks of 16B
  for (int i = tid; i < 512; i += 256) {
    const int r = i >> 3, c = (i & 7) << 3;
    async_copy_b128(&sK[r * 64 + c], &Kh[base + (size_t)(k0 + r) * DEPTH + c]);
  }
  wait_async_all();
  __syncthreads();

  v8f acc[4];
#pragma unroll
  for (int t = 0; t < 4; ++t)
#pragma unroll
    for (int r = 0; r < 8; ++r) acc[t][r] = 0.f;

#pragma unroll
  for (int d0 = 0; d0 < DEPTH; d0 += 32) {
    v16bf a, b0, b1, b2, b3;
    load_a_frag(sQ + wave * 16 * 64 + d0, 64, lane, a);
    load_b_frag(sK + 0 * 16 * 64 + d0, 64, lane, b0);
    load_b_frag(sK + 1 * 16 * 64 + d0, 64, lane, b1);
    load_b_frag(sK + 2 * 16 * 64 + d0, 64, lane, b2);
    load_b_frag(sK + 3 * 16 * 64 + d0, 64, lane, b3);
    acc[0] = wmma_bf16(a, b0, acc[0]);
    acc[1] = wmma_bf16(a, b1, acc[1]);
    acc[2] = wmma_bf16(a, b2, acc[2]);
    acc[3] = wmma_bf16(a, b3, acc[3]);
  }

  const int ncol  = lane & 15;
  const int mbase = (lane >> 4) << 3;
  const float scale = 0.125f;  // 1/sqrt(64)
#pragma unroll
  for (int t = 0; t < 4; ++t) {
    const int kk = k0 + t * 16 + ncol;
    const float mv = (float)mask[b * SS + kk] * -1e9f;
#pragma unroll
    for (int r = 0; r < 8; ++r) {
      const int q = q0 + wave * 16 + mbase + r;
      attn[((size_t)bh * SS + q) * SS + kk] = acc[t][r] * scale + mv;
    }
  }
}

// ---------------------------------------------------------------------------
// Kernel 3: softmax over the HEAD axis (axis=1 of [B,H,S,S]) -- matches ref.
// ---------------------------------------------------------------------------
__global__ __launch_bounds__(256) void head_softmax_kernel(float* __restrict__ attn)
{
  const size_t idx = (size_t)blockIdx.x * 256 + threadIdx.x;  // over B*S*S
  const size_t plane = (size_t)SS * SS;
  const size_t b   = idx / plane;
  const size_t rem = idx - b * plane;  // q*S + k
  float* p = attn + b * (size_t)HH * plane + rem;

  float v[HH];
  float mx = -3.402823466e38f;
#pragma unroll
  for (int h = 0; h < HH; ++h) {
    v[h] = p[(size_t)h * plane];
    mx = fmaxf(mx, v[h]);
  }
  float sum = 0.f;
#pragma unroll
  for (int h = 0; h < HH; ++h) {
    v[h] = __expf(v[h] - mx);
    sum += v[h];
  }
  const float inv = 1.f / sum;
#pragma unroll
  for (int h = 0; h < HH; ++h) p[(size_t)h * plane] = v[h] * inv;
}

// ---------------------------------------------------------------------------
// Kernel 4: ctx[b,q,h,d] = attn[b,h,q,:] @ V  with V stored transposed
// ([B,H,DEPTH,S]) so the B-tile stage is a contiguous async copy.
// ---------------------------------------------------------------------------
__global__ __launch_bounds__(256) void ctx_kernel(
    const float* __restrict__ attn,  // [B,H,S,S] fp32 (post-softmax)
    const __bf16* __restrict__ Vt,   // [B,H,DEPTH,S] bf16 (transposed)
    __bf16* __restrict__ Ctx)        // [B*S, D] merged heads
{
  __shared__ __attribute__((aligned(16))) __bf16 sP[128 * 32];
  __shared__ __attribute__((aligned(16))) __bf16 sV[64 * 32];
  const int tid  = threadIdx.x;
  const int wave = tid >> 5;
  const int lane = tid & 31;
  const int q0 = blockIdx.x * 128;
  const int bh = blockIdx.y;
  const int b = bh >> 3, h = bh & 7;
  const size_t abase = (size_t)bh * SS * SS;
  const size_t vbase = (size_t)bh * DEPTH * SS;

  v8f acc[4];
#pragma unroll
  for (int t = 0; t < 4; ++t)
#pragma unroll
    for (int r = 0; r < 8; ++r) acc[t][r] = 0.f;

  for (int k0 = 0; k0 < SS; k0 += 32) {
    // sV: 64 d-rows x 32 k = 256 chunks of 16B, contiguous in Vt rows (async)
    if (tid < 256) {
      const int n = tid >> 2, c = (tid & 3) << 3;
      async_copy_b128(&sV[n * 32 + c], &Vt[vbase + (size_t)n * SS + k0 + c]);
    }
    // sP: fp32 -> bf16 convert (VALU path)
    for (int i = tid; i < 128 * 32; i += 256) {
      const int r = i >> 5, c = i & 31;
      sP[i] = (__bf16)attn[abase + (size_t)(q0 + r) * SS + k0 + c];
    }
    // prefetch next attn tile rows while we compute
    if (k0 + 32 < SS)
      __builtin_prefetch(&attn[abase + (size_t)(q0 + (tid >> 1)) * SS + k0 + 32 +
                               ((tid & 1) << 4)], 0, 1);
    wait_async_all();
    __syncthreads();
    v16bf a, b0, b1, b2, b3;
    load_a_frag(sP + wave * 16 * 32, 32, lane, a);
    load_b_frag(sV + 0 * 16 * 32, 32, lane, b0);
    load_b_frag(sV + 1 * 16 * 32, 32, lane, b1);
    load_b_frag(sV + 2 * 16 * 32, 32, lane, b2);
    load_b_frag(sV + 3 * 16 * 32, 32, lane, b3);
    acc[0] = wmma_bf16(a, b0, acc[0]);
    acc[1] = wmma_bf16(a, b1, acc[1]);
    acc[2] = wmma_bf16(a, b2, acc[2]);
    acc[3] = wmma_bf16(a, b3, acc[3]);
    __syncthreads();
  }

  const int ncol  = lane & 15;
  const int mbase = (lane >> 4) << 3;
#pragma unroll
  for (int t = 0; t < 4; ++t) {
    const int d = t * 16 + ncol;
#pragma unroll
    for (int r = 0; r < 8; ++r) {
      const int q = q0 + wave * 16 + mbase + r;
      const int m = b * SS + q;
      Ctx[(size_t)m * DD + h * DEPTH + d] = (__bf16)acc[t][r];
    }
  }
}

// ---------------------------------------------------------------------------
// Kernel 5: out = Ctx @ Wo^T + bo  (fp32 out, row-major [B*S, D])
// Ctx tile (already bf16) staged via async copies; W tile converts on VALU.
// ---------------------------------------------------------------------------
__global__ __launch_bounds__(256) void outproj_kernel(
    const __bf16* __restrict__ Ctx,  // [B*S, D] bf16
    const float* __restrict__ W,     // [D, D]
    const float* __restrict__ bias,  // [D]
    float* __restrict__ Out)         // [B*S, D] fp32
{
  __shared__ __attribute__((aligned(16))) __bf16 sA[128 * 32];
  __shared__ __attribute__((aligned(16))) __bf16 sB[64 * 32];
  const int tid  = threadIdx.x;
  const int wave = tid >> 5;
  const int lane = tid & 31;
  const int m0 = blockIdx.x * 128;
  const int n0 = blockIdx.y * 64;

  v8f acc[4];
#pragma unroll
  for (int t = 0; t < 4; ++t)
#pragma unroll
    for (int r = 0; r < 8; ++r) acc[t][r] = 0.f;

  for (int k0 = 0; k0 < DD; k0 += 32) {
    // sA: 128 rows x 32 bf16 = 512 chunks of 16B (async, no conversion)
    for (int i = tid; i < 512; i += 256) {
      const int r = i >> 2, c = (i & 3) << 3;
      async_copy_b128(&sA[r * 32 + c], &Ctx[(size_t)(m0 + r) * DD + k0 + c]);
    }
    for (int i = tid; i < 64 * 32; i += 256) {
      const int n = i >> 5, c = i & 31;
      sB[i] = (__bf16)W[(size_t)(n0 + n) * DD + k0 + c];
    }
    wait_async_all();
    __syncthreads();
    v16bf a, b0, b1, b2, b3;
    load_a_frag(sA + wave * 16 * 32, 32, lane, a);
    load_b_frag(sB + 0 * 16 * 32, 32, lane, b0);
    load_b_frag(sB + 1 * 16 * 32, 32, lane, b1);
    load_b_frag(sB + 2 * 16 * 32, 32, lane, b2);
    load_b_frag(sB + 3 * 16 * 32, 32, lane, b3);
    acc[0] = wmma_bf16(a, b0, acc[0]);
    acc[1] = wmma_bf16(a, b1, acc[1]);
    acc[2] = wmma_bf16(a, b2, acc[2]);
    acc[3] = wmma_bf16(a, b3, acc[3]);
    __syncthreads();
  }

  const int ncol  = lane & 15;
  const int mbase = (lane >> 4) << 3;
#pragma unroll
  for (int t = 0; t < 4; ++t) {
    const int n = n0 + t * 16 + ncol;
    const float bv = bias[n];
#pragma unroll
    for (int r = 0; r < 8; ++r) {
      const int m = m0 + wave * 16 + mbase + r;
      Out[(size_t)m * DD + n] = acc[t][r] + bv;
    }
  }
}

// ---------------------------------------------------------------------------
// Launch
// ---------------------------------------------------------------------------
extern "C" void kernel_launch(void* const* d_in, const int* in_sizes, int n_in,
                              void* d_out, int out_size, void* d_ws, size_t ws_size,
                              hipStream_t stream) {
  const float* q    = (const float*)d_in[0];
  const float* k    = (const float*)d_in[1];
  const float* v    = (const float*)d_in[2];
  const int*   mask = (const int*)d_in[3];
  const float* wq_w = (const float*)d_in[4];
  const float* wq_b = (const float*)d_in[5];
  const float* wk_w = (const float*)d_in[6];
  const float* wk_b = (const float*)d_in[7];
  const float* wv_w = (const float*)d_in[8];
  const float* wv_b = (const float*)d_in[9];
  const float* wo_w = (const float*)d_in[10];
  const float* wo_b = (const float*)d_in[11];

  const size_t proj_elems = (size_t)BB * SS * DD;  // 8,388,608
  __bf16* qh = (__bf16*)d_ws;
  __bf16* kh = qh + proj_elems;
  __bf16* vt = kh + proj_elems;   // transposed V: [B,H,DEPTH,S]
  __bf16* ctx = qh;  // Q region reused after logits; stream order guarantees safety

  float* out  = (float*)d_out;
  float* attn = out + proj_elems;  // [B,H,S,S]

  const dim3 blk(256);
  // 1) QKV projections (bf16 WMMA); stride-parameterized head-split layouts:
  //    Q,K: [b,h,s,d]  (s_stride=DEPTH, d_stride=1)
  //    V:   [b,h,d,s]  (s_stride=1,     d_stride=SS) for async B-staging in PV
  proj_qkv_kernel<<<dim3(BB * SS / 128, DD / 64), blk, 0, stream>>>(
      q, wq_w, wq_b, qh, DEPTH, 1);
  proj_qkv_kernel<<<dim3(BB * SS / 128, DD / 64), blk, 0, stream>>>(
      k, wk_w, wk_b, kh, DEPTH, 1);
  proj_qkv_kernel<<<dim3(BB * SS / 128, DD / 64), blk, 0, stream>>>(
      v, wv_w, wv_b, vt, 1, SS);
  // 2) scaled+masked logits straight into the attn output region
  logits_kernel<<<dim3(SS / 128, SS / 64, BB * HH), blk, 0, stream>>>(qh, kh, mask, attn);
  // 3) softmax over the head axis (faithful to the reference's axis=1)
  head_softmax_kernel<<<dim3((unsigned)((size_t)BB * SS * SS / 256)), blk, 0, stream>>>(attn);
  // 4) context = attn @ V  (merged-head bf16)
  ctx_kernel<<<dim3(SS / 128, BB * HH), blk, 0, stream>>>(attn, vt, ctx);
  // 5) output projection (fp32)
  outproj_kernel<<<dim3(BB * SS / 128, DD / 64), blk, 0, stream>>>(ctx, wo_w, wo_b, out);
}
